// SelfAttention_33097017983523
// MI455X (gfx1250) — matrix operand order repaired and checked by reading, not compiled, not run on hardware
//
#include <hip/hip_runtime.h>
#include <hip/hip_bf16.h>

// ---------------------------------------------------------------------------
// Self-attention (S=2048, HIDDEN=2048, 32 heads x 64, 8 KV heads) for gfx1250.
// All GEMMs use v_wmma_f32_16x16x32_bf16 (wave32, 16x16 tiles, K=32 steps).
// LDS staging uses GLOBAL_LOAD_ASYNC_TO_LDS_B128 (ASYNCcnt-tracked DMA path).
// ---------------------------------------------------------------------------

typedef __bf16 bf16;
typedef __attribute__((ext_vector_type(16))) __bf16 v16bf;
typedef __attribute__((ext_vector_type(8)))  float  v8f;
typedef int v4i __attribute__((ext_vector_type(4)));

union FragBf {
    v16bf v;
    struct { uint4 lo; uint4 hi; } u;
};

__device__ __forceinline__ bf16 to_bf16(float f) {
    union { float f; unsigned u; } x; x.f = f;
    unsigned r = x.u + 0x7FFFu + ((x.u >> 16) & 1u);   // round-to-nearest-even
    union { unsigned short s; bf16 b; } y; y.s = (unsigned short)(r >> 16);
    return y.b;
}

__device__ __forceinline__ v8f wmma_bf16(const FragBf& a, const FragBf& b, v8f c) {
    return __builtin_amdgcn_wmma_f32_16x16x32_bf16(false, a.v, false, b.v,
                                                   (short)0, c, false, false);
}

// ---- async memory->LDS staging (CDNA5 GLOBAL_LOAD_ASYNC_TO_LDS_B128) ------
#if __has_builtin(__builtin_amdgcn_global_load_async_to_lds_b128) && \
    __has_builtin(__builtin_amdgcn_s_wait_asynccnt)
#define HAVE_ASYNC_LDS 1
typedef __attribute__((address_space(1))) v4i g_v4i;   // global int4
typedef __attribute__((address_space(3))) v4i l_v4i;   // LDS int4
__device__ __forceinline__ void stage16(const bf16* __restrict__ g,
                                        bf16* __restrict__ l) {
    __builtin_amdgcn_global_load_async_to_lds_b128(
        (g_v4i*)(void*)g, (l_v4i*)(void*)l, 0, 0);
}
__device__ __forceinline__ void stage_wait() {
    __builtin_amdgcn_s_wait_asynccnt(0);
}
#else
#define HAVE_ASYNC_LDS 0
__device__ __forceinline__ void stage16(const bf16* __restrict__ g,
                                        bf16* __restrict__ l) {
    *(uint4*)l = *(const uint4*)g;
}
__device__ __forceinline__ void stage_wait() {}
#endif

// ---------------------------------------------------------------------------
// fp32 -> bf16 elementwise (for x). n must be a multiple of 4.
// ---------------------------------------------------------------------------
__global__ void cvt_bf16_kernel(const float* __restrict__ in,
                                bf16* __restrict__ out, int n) {
    int i = (blockIdx.x * blockDim.x + threadIdx.x) * 4;
    if (i + 3 >= n) return;
    float4 f = *(const float4*)(in + i);
    union { bf16 b[4]; unsigned long long u; } pk;
    pk.b[0] = to_bf16(f.x); pk.b[1] = to_bf16(f.y);
    pk.b[2] = to_bf16(f.z); pk.b[3] = to_bf16(f.w);
    *(unsigned long long*)(out + i) = pk.u;
}

// ---------------------------------------------------------------------------
// fp32 [R][C] -> bf16 [C][R] (transpose so WMMA B-fragments read contiguous K)
// ---------------------------------------------------------------------------
__global__ void transpose_cvt_kernel(const float* __restrict__ in,
                                     bf16* __restrict__ out, int R, int C) {
    __shared__ float tile[32][33];
    int c0 = blockIdx.x * 32, r0 = blockIdx.y * 32;
    #pragma unroll
    for (int j = threadIdx.y; j < 32; j += 8)
        tile[j][threadIdx.x] = in[(size_t)(r0 + j) * C + c0 + threadIdx.x];
    __syncthreads();
    #pragma unroll
    for (int j = threadIdx.y; j < 32; j += 8)
        out[(size_t)(c0 + j) * R + r0 + threadIdx.x] = to_bf16(tile[threadIdx.x][j]);
}

// ---------------------------------------------------------------------------
// Generic bf16 GEMM: C[M][N] = A[M][K] * Bt[N][K]^T, fp32 accumulate.
// Workgroup tile: 128(M) x 64(N), 8 waves; each wave: 16(M) x 64(N).
// EPI 0: fp32 row-major out.  EPI 1: bf16 transposed out[N][M].
// EPI 2: RoPE epilogue (per-64 head; pairs (d, d+32) are acc tiles t, t+2),
//        then scale, bf16 row-major out.
// ---------------------------------------------------------------------------
template <int EPI>
__global__ __launch_bounds__(256)
void gemm_bf16_kernel(const bf16* __restrict__ A, const bf16* __restrict__ Bt,
                      void* __restrict__ Cout, int M, int N, int K,
                      const float* __restrict__ inv_freq, float scale) {
    constexpr int BM = 128, BN = 64, BK = 32;
    __shared__ __align__(16) bf16 sA[BM * BK];   // [128][32]
    __shared__ __align__(16) bf16 sB[BN * BK];   // [64][32]

    const int tid  = threadIdx.x;
    const int wave = tid >> 5;
    const int lane = tid & 31;
    const int lm   = lane & 15;
    const int half = lane >> 4;
    const int m0 = blockIdx.y * BM;
    const int n0 = blockIdx.x * BN;

    v8f acc[4];
    #pragma unroll
    for (int t = 0; t < 4; ++t)
        #pragma unroll
        for (int r = 0; r < 8; ++r) acc[t][r] = 0.0f;

    const int arow = tid >> 1, achunk = (tid & 1) * 16;      // A stage: 2x16B/thread
    const int brow = tid >> 2, bchunk = (tid & 3) * 8;       // B stage: 1x16B/thread
    const int koff = half * 8;     // A-fragment K offset per 16-lane half
    const int khal = half * 8;     // B-fragment K offset per 16-lane half

    for (int k0 = 0; k0 < K; k0 += BK) {
        __syncthreads();
        stage16(&A[(m0 + arow) * K + k0 + achunk],     &sA[arow * BK + achunk]);
        stage16(&A[(m0 + arow) * K + k0 + achunk + 8], &sA[arow * BK + achunk + 8]);
        stage16(&Bt[(n0 + brow) * K + k0 + bchunk],    &sB[brow * BK + bchunk]);
        if (k0 + BK < K) {   // prefetch next K slice
            __builtin_prefetch(&A[(m0 + arow) * K + k0 + BK + achunk], 0, 1);
            __builtin_prefetch(&Bt[(n0 + brow) * K + k0 + BK + bchunk], 0, 1);
        }
        stage_wait();
        __syncthreads();

        FragBf af;
        const bf16* ap = &sA[(wave * 16 + lm) * BK + koff];
        af.u.lo = *(const uint4*)(ap);
        af.u.hi = *(const uint4*)(ap + 16);
        #pragma unroll
        for (int t = 0; t < 4; ++t) {
            FragBf bfm;
            const bf16* bp = &sB[(t * 16 + lm) * BK + khal];
            bfm.u.lo = *(const uint4*)(bp);
            bfm.u.hi = *(const uint4*)(bp + 16);
            acc[t] = wmma_bf16(af, bfm, acc[t]);
        }
    }

    const int rbase = m0 + wave * 16 + half * 8;   // + r = output row
    if (EPI == 0) {
        float* C = (float*)Cout;
        #pragma unroll
        for (int t = 0; t < 4; ++t)
            #pragma unroll
            for (int r = 0; r < 8; ++r)
                C[(size_t)(rbase + r) * N + n0 + t * 16 + lm] = acc[t][r];
    } else if (EPI == 1) {
        bf16* C = (bf16*)Cout;   // transposed: [N][M]
        #pragma unroll
        for (int t = 0; t < 4; ++t)
            #pragma unroll
            for (int r = 0; r < 8; ++r)
                C[(size_t)(n0 + t * 16 + lm) * M + rbase + r] = to_bf16(acc[t][r]);
    } else {
        bf16* C = (bf16*)Cout;   // RoPE + scale, row-major
        #pragma unroll
        for (int t = 0; t < 2; ++t) {
            float invf = inv_freq[t * 16 + lm];    // i = (col % 64) in [0,32)
            #pragma unroll
            for (int r = 0; r < 8; ++r) {
                int srow = rbase + r;
                float ang = (float)srow * invf;
                float sn = sinf(ang), cs = cosf(ang);
                float hv = acc[t][r], ov = acc[t + 2][r];
                C[(size_t)srow * N + n0 + t * 16 + lm] =
                    to_bf16((hv * cs + ov * sn) * scale);
                C[(size_t)srow * N + n0 + 32 + t * 16 + lm] =
                    to_bf16((hv * sn - ov * cs) * scale);
            }
        }
    }
}

// ---------------------------------------------------------------------------
// Flash attention: one workgroup per (128 query rows, head H of 32).
// Q pre-scaled by 1/sqrt(64) and roped; Kt roped; Vt stored [d][s].
// Wave w owns query rows q0+16w..+16; online softmax over 32-key blocks.
// ---------------------------------------------------------------------------
__global__ __launch_bounds__(256)
void flash_kernel(const bf16* __restrict__ Q,   // [S][2048]
                  const bf16* __restrict__ Kt,  // [S][512]
                  const bf16* __restrict__ Vt,  // [512][S]
                  bf16* __restrict__ O,         // [S][2048]
                  int S) {
    constexpr int BM = 128, BC = 32;
    __shared__ __align__(16) bf16 sK[BC * 64];       // [32 keys][64 d]
    __shared__ __align__(16) bf16 sV[64 * BC];       // [64 d][32 keys]
    __shared__ __align__(16) bf16 sP[8 * 16 * BC];   // per-wave [16 m][32 b]

    const int tid  = threadIdx.x;
    const int wave = tid >> 5;
    const int lane = tid & 31;
    const int lm   = lane & 15;
    const int half = lane >> 4;
    const int H    = blockIdx.y;               // flat head 0..31
    const int q0   = blockIdx.x * BM;
    const int qcol = H * 64;
    const int kcol = (H >> 2) * 64;            // kv-head column in Kt
    const int vrow = (H >> 2) * 64;            // kv-head row block in Vt

    // Preload Q fragments for this wave (K-dim steps 0..31 and 32..63)
    FragBf aq[2];
    {
        const bf16* qp = &Q[(size_t)(q0 + wave * 16 + lm) * 2048 + qcol];
        const int koff = half * 8;
        #pragma unroll
        for (int s = 0; s < 2; ++s) {
            aq[s].u.lo = *(const uint4*)(qp + s * 32 + koff);
            aq[s].u.hi = *(const uint4*)(qp + s * 32 + koff + 16);
        }
    }

    v8f oacc[4];
    #pragma unroll
    for (int t = 0; t < 4; ++t)
        #pragma unroll
        for (int r = 0; r < 8; ++r) oacc[t][r] = 0.0f;
    float mrun[8], lrun[8];
    #pragma unroll
    for (int r = 0; r < 8; ++r) { mrun[r] = -3.0e38f; lrun[r] = 0.0f; }

    const int arow0 = q0 + wave * 16 + half * 8;  // + r = query row for stats
    const int nblocks = (q0 + BM) / BC;           // causal upper bound (uniform)

    // staging indices (256 threads cover both tiles)
    const int kr = tid >> 3, kc = (tid & 7) * 8;  // K tile [32][64]
    const int vr = tid >> 2, vc = (tid & 3) * 8;  // V^T tile [64][32]

    for (int j = 0; j < nblocks; ++j) {
        const int b0 = j * BC;
        __syncthreads();
        stage16(&Kt[(size_t)(b0 + kr) * 512 + kcol + kc], &sK[kr * 64 + kc]);
        stage16(&Vt[(size_t)(vrow + vr) * S + b0 + vc],   &sV[vr * BC + vc]);
        if (j + 1 < nblocks) {   // prefetch next key block
            __builtin_prefetch(&Kt[(size_t)(b0 + BC + kr) * 512 + kcol + kc], 0, 1);
            __builtin_prefetch(&Vt[(size_t)(vrow + vr) * S + b0 + BC + vc], 0, 1);
        }
        stage_wait();
        __syncthreads();

        // ---- S = Q K^T (two 16-wide key tiles, two K=32 steps) ----
        v8f sacc[2];
        #pragma unroll
        for (int nt = 0; nt < 2; ++nt) {
            v8f a;
            #pragma unroll
            for (int r = 0; r < 8; ++r) a[r] = 0.0f;
            #pragma unroll
            for (int ks = 0; ks < 2; ++ks) {
                FragBf bk;
                const bf16* bp = &sK[(nt * 16 + lm) * 64 + ks * 32 + half * 8];
                bk.u.lo = *(const uint4*)(bp);
                bk.u.hi = *(const uint4*)(bp + 16);
                a = wmma_bf16(aq[ks], bk, a);
            }
            sacc[nt] = a;
        }

        // ---- causal mask + online softmax ----
        float mblk[8];
        #pragma unroll
        for (int r = 0; r < 8; ++r) mblk[r] = -3.0e38f;
        #pragma unroll
        for (int nt = 0; nt < 2; ++nt) {
            int bcol = b0 + nt * 16 + lm;
            #pragma unroll
            for (int r = 0; r < 8; ++r) {
                float s = sacc[nt][r];
                if (bcol > arow0 + r) s = -3.0e38f;
                sacc[nt][r] = s;
                mblk[r] = fmaxf(mblk[r], s);
            }
        }
        #pragma unroll
        for (int off = 1; off < 16; off <<= 1)
            #pragma unroll
            for (int r = 0; r < 8; ++r)
                mblk[r] = fmaxf(mblk[r], __shfl_xor(mblk[r], off, 32));
        float pscale[8];
        #pragma unroll
        for (int r = 0; r < 8; ++r) {
            float mnew = fmaxf(mrun[r], mblk[r]);
            pscale[r] = __expf(mrun[r] - mnew);
            mrun[r] = mnew;
        }
        float rsum[8];
        #pragma unroll
        for (int r = 0; r < 8; ++r) rsum[r] = 0.0f;
        #pragma unroll
        for (int nt = 0; nt < 2; ++nt)
            #pragma unroll
            for (int r = 0; r < 8; ++r) {
                float p = __expf(sacc[nt][r] - mrun[r]);
                sacc[nt][r] = p;
                rsum[r] += p;
            }
        #pragma unroll
        for (int off = 1; off < 16; off <<= 1)
            #pragma unroll
            for (int r = 0; r < 8; ++r)
                rsum[r] += __shfl_xor(rsum[r], off, 32);
        #pragma unroll
        for (int r = 0; r < 8; ++r) lrun[r] = lrun[r] * pscale[r] + rsum[r];
        #pragma unroll
        for (int t = 0; t < 4; ++t)
            #pragma unroll
            for (int r = 0; r < 8; ++r) oacc[t][r] *= pscale[r];

        // ---- route P through LDS into A-fragment layout ----
        bf16* pw = &sP[wave * 16 * BC];
        #pragma unroll
        for (int nt = 0; nt < 2; ++nt)
            #pragma unroll
            for (int r = 0; r < 8; ++r)
                pw[(r + half * 8) * BC + nt * 16 + lm] = to_bf16(sacc[nt][r]);
        __syncthreads();
        FragBf ap;
        {
            const bf16* p = &pw[lm * BC + half * 8];
            ap.u.lo = *(const uint4*)(p);
            ap.u.hi = *(const uint4*)(p + 16);
        }
        // ---- O += P V (four 16-wide d tiles, K=32 keys) ----
        #pragma unroll
        for (int t = 0; t < 4; ++t) {
            FragBf bv;
            const bf16* bp = &sV[(t * 16 + lm) * BC + half * 8];
            bv.u.lo = *(const uint4*)(bp);
            bv.u.hi = *(const uint4*)(bp + 16);
            oacc[t] = wmma_bf16(ap, bv, oacc[t]);
        }
    }

    // ---- normalize and store ----
    #pragma unroll
    for (int t = 0; t < 4; ++t)
        #pragma unroll
        for (int r = 0; r < 8; ++r)
            O[(size_t)(arow0 + r) * 2048 + qcol + t * 16 + lm] =
                to_bf16(oacc[t][r] / lrun[r]);
}

// ---------------------------------------------------------------------------
// Host-side orchestration
// ---------------------------------------------------------------------------
extern "C" void kernel_launch(void* const* d_in, const int* in_sizes, int n_in,
                              void* d_out, int out_size, void* d_ws, size_t ws_size,
                              hipStream_t stream) {
    const float* x        = (const float*)d_in[0];
    const float* wq       = (const float*)d_in[1];
    const float* wk       = (const float*)d_in[2];
    const float* wv       = (const float*)d_in[3];
    const float* wo       = (const float*)d_in[4];
    const float* inv_freq = (const float*)d_in[5];
    float* out = (float*)d_out;

    const int S = 2048, HID = 2048, KVW = 512;

    char* ws = (char*)d_ws;
    size_t off = 0;
    auto alloc = [&](size_t bytes) {
        void* p = ws + off;
        off += (bytes + 255) & ~(size_t)255;
        return p;
    };
    bf16* xb  = (bf16*)alloc((size_t)S * HID * 2);
    bf16* wqT = (bf16*)alloc((size_t)HID * HID * 2);
    bf16* wkT = (bf16*)alloc((size_t)HID * KVW * 2);
    bf16* wvT = (bf16*)alloc((size_t)HID * KVW * 2);
    bf16* woT = (bf16*)alloc((size_t)HID * HID * 2);
    bf16* qb  = (bf16*)alloc((size_t)S * HID * 2);
    bf16* kb  = (bf16*)alloc((size_t)S * KVW * 2);
    bf16* vbT = (bf16*)alloc((size_t)KVW * S * 2);
    bf16* ob  = (bf16*)alloc((size_t)S * HID * 2);

    // 1) precision conversion (+ weight transpose for contiguous B-fragments)
    cvt_bf16_kernel<<<(S * HID / 4 + 255) / 256, 256, 0, stream>>>(x, xb, S * HID);
    transpose_cvt_kernel<<<dim3(HID / 32, HID / 32), dim3(32, 8), 0, stream>>>(wq, wqT, HID, HID);
    transpose_cvt_kernel<<<dim3(KVW / 32, HID / 32), dim3(32, 8), 0, stream>>>(wk, wkT, HID, KVW);
    transpose_cvt_kernel<<<dim3(KVW / 32, HID / 32), dim3(32, 8), 0, stream>>>(wv, wvT, HID, KVW);
    transpose_cvt_kernel<<<dim3(HID / 32, HID / 32), dim3(32, 8), 0, stream>>>(wo, woT, HID, HID);

    // 2) projections (RoPE fused into Q/K epilogue; 1/sqrt(64) folded into Q;
    //    V written transposed for the PV matmul)
    gemm_bf16_kernel<2><<<dim3(HID / 64, S / 128), 256, 0, stream>>>(
        xb, wqT, qb, S, HID, HID, inv_freq, 0.125f);
    gemm_bf16_kernel<2><<<dim3(KVW / 64, S / 128), 256, 0, stream>>>(
        xb, wkT, kb, S, KVW, HID, inv_freq, 1.0f);
    gemm_bf16_kernel<1><<<dim3(KVW / 64, S / 128), 256, 0, stream>>>(
        xb, wvT, vbT, S, KVW, HID, nullptr, 1.0f);

    // 3) causal flash attention
    flash_kernel<<<dim3(S / 128, 32), 256, 0, stream>>>(qb, kb, vbT, ob, S);

    // 4) output projection -> fp32 result
    gemm_bf16_kernel<0><<<dim3(HID / 64, S / 128), 256, 0, stream>>>(
        ob, woT, out, S, HID, HID, nullptr, 1.0f);
}
